// AttentionHead_6863357739799
// MI455X (gfx1250) — matrix operand order repaired and checked by reading, not compiled, not run on hardware
//
#include <hip/hip_runtime.h>
#include <hip/hip_bf16.h>

typedef __attribute__((ext_vector_type(16))) _Float16 v16h;
typedef __attribute__((ext_vector_type(8)))  float    v8f;

union AF { v16h v; uint32_t u[8]; };
union H2 { uint32_t u; _Float16 h[2]; };

// A-fragment (16x32 f16): lane L -> row m=L%16, half h=L/16.
// VGPR v holds the f16 pair at k = kA(v,h) = 16*(v>>2) + 2*(v&3) + 8*h.
// => v=0..3 covers halves [8h .. 8h+7], v=4..7 covers [16+8h .. 16+8h+7].
// B-fragment (32x16 f16): lane L -> col n=L%16, half h=L/16.
// VGPR v holds the f16 pair at k = kA(v,h) like A (qkv_kernel instead uses k = 16h + 2v for A and B).

__device__ __forceinline__ void unpack4(AF& f, int base, uint4 q) {
    f.u[base + 0] = q.x; f.u[base + 1] = q.y; f.u[base + 2] = q.z; f.u[base + 3] = q.w;
}

// ---------------------------------------------------------------------------
// Kernel 1: pack W{q,k,v} (fp32 [512][64]) into half2 k-pairs, TRANSPOSED:
// Wpkt[m][n][kp] so a lane's 8 B-fragment dwords are 32 contiguous bytes.
// ---------------------------------------------------------------------------
__global__ void pack_w_kernel(const float* __restrict__ Wq,
                              const float* __restrict__ Wk,
                              const float* __restrict__ Wv,
                              uint32_t* __restrict__ Wpkt) {
    int idx = blockIdx.x * blockDim.x + threadIdx.x;
    if (idx >= 3 * 64 * 256) return;
    int m  = idx / (64 * 256);
    int r  = idx % (64 * 256);
    int n  = r / 256;
    int kp = r % 256;
    const float* W = (m == 0) ? Wq : ((m == 1) ? Wk : Wv);
    H2 p;
    p.h[0] = (_Float16)W[(2 * kp + 0) * 64 + n];
    p.h[1] = (_Float16)W[(2 * kp + 1) * 64 + n];
    Wpkt[idx] = p.u;   // [m][n][kp]
}

// ---------------------------------------------------------------------------
// Kernel 2: QKV projection. M=16384, N=64, K=512, f16 WMMA, f32 accumulate.
// Each wave owns a 16-row tile; 12 accumulators (3 matrices x 4 N-tiles).
// Outputs: Qh (row-major f16, pre-scaled by 1/8), Kh (row-major f16),
//          Vt (f16 transposed per batch: [B][64][4096]).
// ---------------------------------------------------------------------------
__global__ void __launch_bounds__(128) qkv_kernel(
    const float* __restrict__ x, const uint32_t* __restrict__ Wpkt,
    const float* __restrict__ bq, const float* __restrict__ bk,
    const float* __restrict__ bv,
    _Float16* __restrict__ Qh, _Float16* __restrict__ Kh,
    _Float16* __restrict__ Vt) {
    const int lane = threadIdx.x & 31;
    const int w    = threadIdx.x >> 5;
    const int tile = blockIdx.x * 4 + w;        // 0..1023
    const int row0 = tile * 16;
    const int n16  = lane & 15;
    const int h    = lane >> 4;

    v8f acc[3][4];
#pragma unroll
    for (int m = 0; m < 3; ++m)
#pragma unroll
        for (int nt = 0; nt < 4; ++nt) acc[m][nt] = (v8f){};

    const float* xrow = &x[(size_t)(row0 + n16) * 512];

    for (int ks = 0; ks < 16; ++ks) {
        // A fragment: 2 groups of 8 consecutive floats -> cvt_pk into f16 pairs
        AF a;
#pragma unroll
        for (int g = 0; g < 2; ++g) {
            const float4* xv = (const float4*)&xrow[ks * 32 + 16 * h + g * 8];
            float4 f0 = xv[0], f1 = xv[1];
            H2 p;
            p.h[0] = (_Float16)f0.x; p.h[1] = (_Float16)f0.y; a.u[g * 4 + 0] = p.u;
            p.h[0] = (_Float16)f0.z; p.h[1] = (_Float16)f0.w; a.u[g * 4 + 1] = p.u;
            p.h[0] = (_Float16)f1.x; p.h[1] = (_Float16)f1.y; a.u[g * 4 + 2] = p.u;
            p.h[0] = (_Float16)f1.z; p.h[1] = (_Float16)f1.w; a.u[g * 4 + 3] = p.u;
        }
#pragma unroll
        for (int m = 0; m < 3; ++m) {
#pragma unroll
            for (int nt = 0; nt < 4; ++nt) {
                // lane's 8 dwords: kp = ks*16 + 8h + v, contiguous in Wpkt
                const uint4* wv = (const uint4*)&Wpkt[(size_t)m * 16384 +
                                                      (size_t)(nt * 16 + n16) * 256 +
                                                      ks * 16 + 8 * h];
                AF bfr;
                unpack4(bfr, 0, wv[0]);
                unpack4(bfr, 4, wv[1]);
                acc[m][nt] = __builtin_amdgcn_wmma_f32_16x16x32_f16(
                    false, a.v, false, bfr.v, (short)0, acc[m][nt], false, false);
            }
        }
    }

    // Epilogue: bias add + stores (C/D layout: lane=col, VGPR r = row r+8h)
    const int bb = row0 >> 12;            // batch (tiles never straddle batches)
    const int s0 = (row0 & 4095) + 8 * h; // first of 8 consecutive seq positions
#pragma unroll
    for (int nt = 0; nt < 4; ++nt) {
        int col = nt * 16 + n16;
        float biasq = bq[col], biask = bk[col], biasv = bv[col];
#pragma unroll
        for (int r = 0; r < 8; ++r) {
            int row = row0 + r + 8 * h;
            Qh[(size_t)row * 64 + col] = (_Float16)((acc[0][nt][r] + biasq) * 0.125f);
            Kh[(size_t)row * 64 + col] = (_Float16)(acc[1][nt][r] + biask);
        }
        // V transposed: lane's 8 values are consecutive s -> one b128 store
        uint4 vpk;
        H2 p;
        p.h[0] = (_Float16)(acc[2][nt][0] + biasv); p.h[1] = (_Float16)(acc[2][nt][1] + biasv); vpk.x = p.u;
        p.h[0] = (_Float16)(acc[2][nt][2] + biasv); p.h[1] = (_Float16)(acc[2][nt][3] + biasv); vpk.y = p.u;
        p.h[0] = (_Float16)(acc[2][nt][4] + biasv); p.h[1] = (_Float16)(acc[2][nt][5] + biasv); vpk.z = p.u;
        p.h[0] = (_Float16)(acc[2][nt][6] + biasv); p.h[1] = (_Float16)(acc[2][nt][7] + biasv); vpk.w = p.u;
        *(uint4*)&Vt[((size_t)bb * 64 + col) * 4096 + s0] = vpk;
    }
}

// ---------------------------------------------------------------------------
// Kernel 3: fused flash attention. One wave per 16 query rows, 128 chunks of
// 32 keys. Per chunk: 4 WMMA (scores, K=64) + online softmax + 4 WMMA (P*V).
// All fragment loads are b128.
// ---------------------------------------------------------------------------
__global__ void __launch_bounds__(128) attn_kernel(
    const _Float16* __restrict__ Qh, const _Float16* __restrict__ Kh,
    const _Float16* __restrict__ Vt, float* __restrict__ out) {
    __shared__ _Float16 Plds[4][16 * 32];   // per-wave 16x32 P staging tile

    const int lane = threadIdx.x & 31;
    const int w    = threadIdx.x >> 5;
    const int unit = blockIdx.x * 4 + w;    // 0..1023
    const int b    = unit >> 8;             // batch
    const int tile = unit & 255;            // 16-row tile within batch
    const int n16  = lane & 15;
    const int h    = lane >> 4;
    const size_t qrow0 = (size_t)b * 4096 + (size_t)tile * 16;

    // Persistent Q A-fragments (K=64 -> two 16x32 fragments). Q pre-scaled.
    AF aQ[2];
    {
        const _Float16* qp = &Qh[(qrow0 + n16) * 64];
#pragma unroll
        for (int s = 0; s < 2; ++s) {
            unpack4(aQ[s], 0, *(const uint4*)&qp[s * 32 + 8 * h]);
            unpack4(aQ[s], 4, *(const uint4*)&qp[s * 32 + 16 + 8 * h]);
        }
    }

    v8f acc[4];
#pragma unroll
    for (int nt = 0; nt < 4; ++nt) acc[nt] = (v8f){};
    float m_i[8], l_i[8];
#pragma unroll
    for (int r = 0; r < 8; ++r) { m_i[r] = -1e30f; l_i[r] = 0.0f; }

    const _Float16* Kbase = &Kh[(size_t)b * 4096 * 64];
    const _Float16* Vbase = &Vt[(size_t)b * 64 * 4096];

    for (int chunk = 0; chunk < 128; ++chunk) {
        const int j0 = chunk * 32;
        if (chunk + 1 < 128) {  // pull next K/V chunk toward the caches
            __builtin_prefetch(&Kbase[(size_t)(j0 + 32 + lane) * 64], 0, 1);
            __builtin_prefetch(&Vbase[(size_t)(lane * 2) * 4096 + j0 + 32], 0, 1);
        }
        // ---- scores: two 16x16 tiles (keys j0..j0+15, j0+16..j0+31) ----
        v8f st[2];
#pragma unroll
        for (int t = 0; t < 2; ++t) {
            v8f c = (v8f){};
            const _Float16* kp = &Kbase[(size_t)(j0 + t * 16 + n16) * 64 + 8 * h];
#pragma unroll
            for (int s = 0; s < 2; ++s) {
                AF bK;
                unpack4(bK, 0, *(const uint4*)&kp[s * 32]);
                unpack4(bK, 4, *(const uint4*)&kp[s * 32 + 16]);
                c = __builtin_amdgcn_wmma_f32_16x16x32_f16(
                    false, aQ[s].v, false, bK.v, (short)0, c, false, false);
            }
            st[t] = c;
        }
        // ---- online softmax (rows live in VGPR index, cols across lanes) ----
#pragma unroll
        for (int r = 0; r < 8; ++r) {
            float x0 = st[0][r], x1 = st[1][r];
            float mx = fmaxf(x0, x1);
            mx = fmaxf(mx, __shfl_xor(mx, 1, 32));
            mx = fmaxf(mx, __shfl_xor(mx, 2, 32));
            mx = fmaxf(mx, __shfl_xor(mx, 4, 32));
            mx = fmaxf(mx, __shfl_xor(mx, 8, 32));
            float mn = fmaxf(m_i[r], mx);
            float al = __expf(m_i[r] - mn);
            float p0 = __expf(x0 - mn);
            float p1 = __expf(x1 - mn);
            float sum = p0 + p1;
            sum += __shfl_xor(sum, 1, 32);
            sum += __shfl_xor(sum, 2, 32);
            sum += __shfl_xor(sum, 4, 32);
            sum += __shfl_xor(sum, 8, 32);
            l_i[r] = l_i[r] * al + sum;
            m_i[r] = mn;
            acc[0][r] *= al; acc[1][r] *= al; acc[2][r] *= al; acc[3][r] *= al;
            Plds[w][(r + 8 * h) * 32 + n16]      = (_Float16)p0;
            Plds[w][(r + 8 * h) * 32 + 16 + n16] = (_Float16)p1;
        }
        asm volatile("s_wait_dscnt 0x0" ::: "memory");  // P RAW through LDS
        // ---- P A-fragment from LDS (2x ds_load_b128) ----
        AF aP;
        {
            const _Float16* pp = &Plds[w][n16 * 32 + 8 * h];
            unpack4(aP, 0, *(const uint4*)&pp[0]);
            unpack4(aP, 4, *(const uint4*)&pp[16]);
        }
        // ---- acc += P (16x32) x V (32x64) ----
#pragma unroll
        for (int nt = 0; nt < 4; ++nt) {
            const _Float16* vp = &Vbase[(size_t)(nt * 16 + n16) * 4096 + j0 + 8 * h];
            AF bV;
            unpack4(bV, 0, *(const uint4*)&vp[0]);
            unpack4(bV, 4, *(const uint4*)&vp[16]);
            acc[nt] = __builtin_amdgcn_wmma_f32_16x16x32_f16(
                false, aP.v, false, bV.v, (short)0, acc[nt], false, false);
        }
    }

    // ---- epilogue: normalize by row sums, fp32 store ----
#pragma unroll
    for (int nt = 0; nt < 4; ++nt) {
#pragma unroll
        for (int r = 0; r < 8; ++r) {
            size_t row = qrow0 + r + 8 * h;
            out[row * 64 + nt * 16 + n16] = acc[nt][r] / l_i[r];
        }
    }
}

// ---------------------------------------------------------------------------
extern "C" void kernel_launch(void* const* d_in, const int* in_sizes, int n_in,
                              void* d_out, int out_size, void* d_ws, size_t ws_size,
                              hipStream_t stream) {
    (void)in_sizes; (void)n_in; (void)out_size; (void)ws_size;
    const float* x  = (const float*)d_in[0];
    const float* Wq = (const float*)d_in[1];
    const float* bq = (const float*)d_in[2];
    const float* Wk = (const float*)d_in[3];
    const float* bk = (const float*)d_in[4];
    const float* Wv = (const float*)d_in[5];
    const float* bv = (const float*)d_in[6];
    float* out = (float*)d_out;

    char* ws = (char*)d_ws;
    _Float16* Qh   = (_Float16*)(ws + 0);                       // 2 MB
    _Float16* Kh   = (_Float16*)(ws + (size_t)2 * 1024 * 1024); // 2 MB
    _Float16* Vt   = (_Float16*)(ws + (size_t)4 * 1024 * 1024); // 2 MB
    uint32_t* Wpkt = (uint32_t*)(ws + (size_t)6 * 1024 * 1024); // 192 KB

    pack_w_kernel<<<192, 256, 0, stream>>>(Wq, Wk, Wv, Wpkt);
    qkv_kernel<<<256, 128, 0, stream>>>(x, Wpkt, bq, bk, bv, Qh, Kh, Vt);
    attn_kernel<<<256, 128, 0, stream>>>(Qh, Kh, Vt, out);
}
